// SysBinder_58926951301704
// MI455X (gfx1250) — matrix operand order repaired and checked by reading, not compiled, not run on hardware
//
#include <hip/hip_runtime.h>
#include <cmath>

// ---------------- CDNA5 WMMA types ----------------
typedef __attribute__((ext_vector_type(16))) __bf16         v16bf;
typedef __attribute__((ext_vector_type(8)))  float          v8f;
typedef __attribute__((ext_vector_type(8)))  unsigned short ushort8;

__device__ __forceinline__ unsigned short f2bf(float f) {
  unsigned int u = __float_as_uint(f);
  u += 0x7fffu + ((u >> 16) & 1u);           // round-to-nearest-even
  return (unsigned short)(u >> 16);
}

// ---------------- generic bf16 TN GEMM via v_wmma_f32_16x16x32_bf16 ----------------
// C[M,N] = alpha * sum_k A[m,k] * Bt[n,k]  (+bias[n]) ; optional relu / accumulate / bf16-out
// Two-level batching: z -> (b1 = z/nb2, b2 = z%nb2), offsets via s*1/s*2 strides.
struct GemmP {
  const unsigned short* A;
  const unsigned short* B;
  void* C;
  const float* bias;
  long lda, ldb, ldc;
  long sA1, sA2, sB1, sB2, sC1, sC2, sb1, sb2;
  int M, N, K, nb2;
  float alpha;
  int flags;
};
enum { GF_TRANSB = 1, GF_RELU = 2, GF_ACC = 4, GF_OUTBF16 = 8 };

template<int WM, int WN, int MF, int NF>
__launch_bounds__(WM * WN * 32)
__global__ void gemm_wmma_k(GemmP p) {
  constexpr int BM = WM * MF * 16, BN = WN * NF * 16, NT = WM * WN * 32;
  constexpr int LDA = 40, LDB = 40;  // padded rows (80B) -> conflict-free b128 LDS reads
  __shared__ __align__(16) unsigned short As[BM * LDA];
  __shared__ __align__(16) unsigned short Bs[BN * LDB];

  const int tid  = threadIdx.x;
  const int lane = tid & 31, wid = tid >> 5;
  const int hsel = lane >> 4, l16 = lane & 15;   // wave32 half-select
  const int wm0 = (wid % WM) * (MF * 16);
  const int wn0 = (wid / WM) * (NF * 16);
  const long m0 = (long)blockIdx.y * BM;
  const long n0 = (long)blockIdx.x * BN;
  const int z = blockIdx.z;
  const int b1 = z / p.nb2, b2 = z % p.nb2;
  const unsigned short* Ag = p.A + b1 * p.sA1 + b2 * p.sA2;
  const unsigned short* Bg = p.B + b1 * p.sB1 + b2 * p.sB2;
  const bool transB = (p.flags & GF_TRANSB) != 0;

  v8f acc[MF][NF] = {};

  for (int k0 = 0; k0 < p.K; k0 += 32) {
    const bool morek = (k0 + 32) < p.K;
    // ---- stage A tile [BM x 32] (16B chunks, coalesced) + prefetch next tile ----
    for (int idx = tid; idx < BM * 4; idx += NT) {
      int r = idx >> 2, h = idx & 3;
      ushort8 val = {0, 0, 0, 0, 0, 0, 0, 0};
      long gm = m0 + r;
      if (gm < p.M) {
        const unsigned short* ap = Ag + gm * p.lda + k0 + h * 8;
        val = *(const ushort8*)ap;
        if (morek) __builtin_prefetch(ap + 32, 0, 0);   // global_prefetch_b8: next K-tile
      }
      *(ushort8*)(&As[r * LDA + h * 8]) = val;
    }
    // ---- stage Bt tile [BN x 32] ----
    if (!transB) {
      for (int idx = tid; idx < BN * 4; idx += NT) {
        int r = idx >> 2, h = idx & 3;
        ushort8 val = {0, 0, 0, 0, 0, 0, 0, 0};
        long gn = n0 + r;
        if (gn < p.N) {
          const unsigned short* bp = Bg + gn * p.ldb + k0 + h * 8;
          val = *(const ushort8*)bp;
          if (morek) __builtin_prefetch(bp + 32, 0, 0);
        }
        *(ushort8*)(&Bs[r * LDB + h * 8]) = val;
      }
    } else {  // B stored [K][N]: transpose while staging
      for (int idx = tid; idx < BN * 32; idx += NT) {
        int nn = idx % BN, kk = idx / BN;
        unsigned short val = 0;
        long gn = n0 + nn;
        if (gn < p.N) {
          const unsigned short* bp = Bg + (long)(k0 + kk) * p.ldb + gn;
          val = *bp;
          if (morek && kk < 2) __builtin_prefetch(bp + 32 * p.ldb, 0, 0);
        }
        Bs[nn * LDB + kk] = val;
      }
    }
    __syncthreads();

    union U { v16bf v; ushort8 h8[2]; };
    v16bf aF[MF], bF[NF];
    // A fragment: lane row M=l16; K chunks {8h..8h+7} and {16+8h..}
    #pragma unroll
    for (int i = 0; i < MF; i++) {
      U u;
      int r = wm0 + i * 16 + l16;
      u.h8[0] = *(const ushort8*)(&As[r * LDA + hsel * 8]);
      u.h8[1] = *(const ushort8*)(&As[r * LDA + 16 + hsel * 8]);
      aF[i] = u.v;
    }
    // B fragment: lane col N=l16; K chunks {8h..8h+7} and {16+8h..}
    #pragma unroll
    for (int j = 0; j < NF; j++) {
      U u;
      int c = wn0 + j * 16 + l16;
      u.h8[0] = *(const ushort8*)(&Bs[c * LDB + hsel * 8]);
      u.h8[1] = *(const ushort8*)(&Bs[c * LDB + 16 + hsel * 8]);
      bF[j] = u.v;
    }
    #pragma unroll
    for (int i = 0; i < MF; i++)
      #pragma unroll
      for (int j = 0; j < NF; j++)
        acc[i][j] = __builtin_amdgcn_wmma_f32_16x16x32_bf16(
            false, aF[i], false, bF[j], (short)0, acc[i][j], false, false);
    __syncthreads();
  }

  // ---- epilogue: element e of v8f -> row M = e + 8*hsel, col N = l16 ----
  const float* biasp = p.bias ? (p.bias + b1 * p.sb1 + b2 * p.sb2) : nullptr;
  const long cOff = b1 * p.sC1 + b2 * p.sC2;
  float* Cf = (float*)p.C;
  unsigned short* Cb = (unsigned short*)p.C;
  #pragma unroll
  for (int i = 0; i < MF; i++) {
    #pragma unroll
    for (int j = 0; j < NF; j++) {
      long col = n0 + wn0 + j * 16 + l16;
      if (col >= p.N) continue;
      float bb = biasp ? biasp[col] : 0.f;
      #pragma unroll
      for (int e = 0; e < 8; e++) {
        long row = m0 + wm0 + i * 16 + hsel * 8 + e;
        if (row >= p.M) continue;
        float val = p.alpha * acc[i][j][e] + bb;
        if (p.flags & GF_RELU) val = fmaxf(val, 0.f);
        long off = cOff + row * p.ldc + col;
        if (p.flags & GF_OUTBF16)  Cb[off] = f2bf(val);
        else if (p.flags & GF_ACC) Cf[off] += val;
        else                       Cf[off] = val;
      }
    }
  }
}

// ---------------- elementwise / reduction kernels ----------------

// LayerNorm over rows of length L (optional affine), fp32 in -> bf16 out.
// Also serves BlockLayerNorm (L=128, g=b=null).
__global__ void ln_rows_k(const float* __restrict__ x, unsigned short* __restrict__ y,
                          const float* __restrict__ g, const float* __restrict__ b, int L) {
  __shared__ float red[256];
  __shared__ float stats[2];
  const int tid = threadIdx.x;
  const long row = blockIdx.x;
  const float* xr = x + row * (long)L;
  float s = 0.f, s2 = 0.f;
  for (int i = tid; i < L; i += 256) { float v = xr[i]; s += v; s2 += v * v; }
  red[tid] = s; __syncthreads();
  for (int o = 128; o; o >>= 1) { if (tid < o) red[tid] += red[tid + o]; __syncthreads(); }
  if (tid == 0) stats[0] = red[0] / L;
  __syncthreads();
  red[tid] = s2; __syncthreads();
  for (int o = 128; o; o >>= 1) { if (tid < o) red[tid] += red[tid + o]; __syncthreads(); }
  if (tid == 0) {
    float m = stats[0];
    stats[1] = rsqrtf(fmaxf(red[0] / L - m * m, 0.f) + 1e-5f);
  }
  __syncthreads();
  float m = stats[0], rstd = stats[1];
  unsigned short* yr = y + row * (long)L;
  for (int i = tid; i < L; i += 256) {
    float v = (xr[i] - m) * rstd;
    if (g) v = v * g[i] + b[i];
    yr[i] = f2bf(v);
  }
}

__global__ void slots_init_k(const float* __restrict__ mu, const float* __restrict__ ls,
                             const float* __restrict__ noise, float* __restrict__ slots, long total) {
  long idx = (long)blockIdx.x * 256 + threadIdx.x;
  if (idx >= total) return;
  int d = idx & 2047;
  slots[idx] = mu[d] + expf(ls[d]) * noise[idx];
}

__global__ void cvt_bf16_k(const float* __restrict__ s, unsigned short* __restrict__ d, long n) {
  long idx = (long)blockIdx.x * 256 + threadIdx.x;
  if (idx < n) d[idx] = f2bf(s[idx]);
}

// dst[z][c][r] = src[z][r][c] (fp32 -> bf16), for weight pre-transposition
__global__ void transpose_cvt_k(const float* __restrict__ s, unsigned short* __restrict__ d,
                                int R, int C, long total) {
  long idx = (long)blockIdx.x * 256 + threadIdx.x;
  if (idx >= total) return;
  long rc = (long)R * C;
  long z = idx / rc, rem = idx - z * rc;
  long c = rem / R, r = rem - c * R;
  d[idx] = f2bf(s[z * rc + r * C + c]);
}

// softmax of dots[b][1024][8] over n (axis=1), + EPS renorm (reference semantics).
// writes attn fp32 [b][n][s] (output) and attn^T bf16 [b][s][n] (for updates GEMM).
__global__ void col_softmax_k(const float* __restrict__ dots, float* __restrict__ attn_out,
                              unsigned short* __restrict__ attn_t) {
  __shared__ float red[256];
  const int tid = threadIdx.x;
  const int b = blockIdx.x >> 3, s = blockIdx.x & 7;
  const float* d = dots + (long)b * 8192 + s;
  float loc[4];
  float m = -1e30f;
  #pragma unroll
  for (int i = 0; i < 4; i++) { loc[i] = d[(tid + i * 256) * 8]; m = fmaxf(m, loc[i]); }
  red[tid] = m; __syncthreads();
  for (int o = 128; o; o >>= 1) { if (tid < o) red[tid] = fmaxf(red[tid], red[tid + o]); __syncthreads(); }
  m = red[0]; __syncthreads();
  float sum = 0.f;
  #pragma unroll
  for (int i = 0; i < 4; i++) { loc[i] = expf(loc[i] - m); sum += loc[i]; }
  red[tid] = sum; __syncthreads();
  for (int o = 128; o; o >>= 1) { if (tid < o) red[tid] += red[tid + o]; __syncthreads(); }
  float Z = red[0]; __syncthreads();
  float s2 = 0.f;
  #pragma unroll
  for (int i = 0; i < 4; i++) { loc[i] = loc[i] / Z + 1e-8f; s2 += loc[i]; }
  red[tid] = s2; __syncthreads();
  for (int o = 128; o; o >>= 1) { if (tid < o) red[tid] += red[tid + o]; __syncthreads(); }
  float S2 = red[0];
  #pragma unroll
  for (int i = 0; i < 4; i++) {
    int n = tid + i * 256;
    float w = loc[i] / S2;
    attn_out[(long)b * 8192 + (long)n * 8 + s] = w;
    attn_t[(long)b * 8192 + (long)s * 1024 + n] = f2bf(w);
  }
}

// softmax over rows of 64 (block-wise attention scores); one wave per row; bf16 out.
__global__ void softmax64_k(const float* __restrict__ sc, unsigned short* __restrict__ out) {
  const int lane = threadIdx.x & 31;
  const long row = (long)blockIdx.x * 8 + (threadIdx.x >> 5);
  const float* r = sc + row * 64;
  float a0 = r[lane], a1 = r[lane + 32];
  float m = fmaxf(a0, a1);
  for (int o = 16; o; o >>= 1) m = fmaxf(m, __shfl_xor(m, o));
  float e0 = expf(a0 - m), e1 = expf(a1 - m);
  float s = e0 + e1;
  for (int o = 16; o; o >>= 1) s += __shfl_xor(s, o);
  out[row * 64 + lane]      = f2bf(e0 / s);
  out[row * 64 + lane + 32] = f2bf(e1 / s);
}

// GRU gate math: gi/gh are [256 rows][16 blk][384]; slots updated in place [256][2048]
__global__ void gru_elem_k(const float* __restrict__ gi, const float* __restrict__ gh,
                           float* __restrict__ slots) {
  long idx = (long)blockIdx.x * 256 + threadIdx.x;  // 524288 total
  int j = idx & 127;
  int blk = (int)((idx >> 7) & 15);
  long n = idx >> 11;
  const float* gir = gi + n * 6144 + blk * 384;
  const float* ghr = gh + n * 6144 + blk * 384;
  float ir = gir[j],        hr = ghr[j];
  float iz = gir[128 + j],  hz = ghr[128 + j];
  float in_ = gir[256 + j], hn = ghr[256 + j];
  float h = slots[idx];
  float r = 1.f / (1.f + expf(-(ir + hr)));
  float zz = 1.f / (1.f + expf(-(iz + hz)));
  float nn = tanhf(in_ + r * hn);
  slots[idx] = (1.f - zz) * nn + zz * h;
}

// ---------------- host side ----------------
static GemmP mkgemm(const unsigned short* A, long lda, long sA1, long sA2,
                    const unsigned short* B, long ldb, long sB1, long sB2,
                    void* C, long ldc, long sC1, long sC2,
                    const float* bias, long sb1, long sb2,
                    int M, int N, int K, int nb2, float alpha, int flags) {
  GemmP p;
  p.A = A; p.B = B; p.C = C; p.bias = bias;
  p.lda = lda; p.ldb = ldb; p.ldc = ldc;
  p.sA1 = sA1; p.sA2 = sA2; p.sB1 = sB1; p.sB2 = sB2; p.sC1 = sC1; p.sC2 = sC2;
  p.sb1 = sb1; p.sb2 = sb2;
  p.M = M; p.N = N; p.K = K; p.nb2 = nb2; p.alpha = alpha; p.flags = flags;
  return p;
}

extern "C" void kernel_launch(void* const* d_in, const int* in_sizes, int n_in,
                              void* d_out, int out_size, void* d_ws, size_t ws_size,
                              hipStream_t stream) {
  (void)in_sizes; (void)n_in; (void)out_size; (void)ws_size;
  const float* inputs = (const float*)d_in[0];
  const float* noise  = (const float*)d_in[1];
  const float* mu     = (const float*)d_in[2];
  const float* lsig   = (const float*)d_in[3];
  const float* Wq  = (const float*)d_in[4];  const float* bq  = (const float*)d_in[5];
  const float* Wk  = (const float*)d_in[6];  const float* bk  = (const float*)d_in[7];
  const float* Wv  = (const float*)d_in[8];  const float* bv  = (const float*)d_in[9];
  const float* gWih = (const float*)d_in[10]; const float* gWhh = (const float*)d_in[11];
  const float* gbih = (const float*)d_in[12]; const float* gbhh = (const float*)d_in[13];
  const float* mW1 = (const float*)d_in[14]; const float* mb1 = (const float*)d_in[15];
  const float* mW2 = (const float*)d_in[16]; const float* mb2 = (const float*)d_in[17];
  const float* cvec = (const float*)d_in[18];
  const float* cW1 = (const float*)d_in[19]; const float* cb1 = (const float*)d_in[20];
  const float* cW2 = (const float*)d_in[21]; const float* cb2 = (const float*)d_in[22];
  const float* cW3 = (const float*)d_in[23]; const float* cb3 = (const float*)d_in[24];
  const float* cW4 = (const float*)d_in[25]; const float* cb4 = (const float*)d_in[26];
  const float* nig = (const float*)d_in[27]; const float* nib = (const float*)d_in[28];
  const float* nsg = (const float*)d_in[29]; const float* nsb = (const float*)d_in[30];

  // workspace bump allocator (256B aligned)
  char* wp = (char*)d_ws;
  auto allocB = [&](size_t n) { unsigned short* r = (unsigned short*)wp; wp += ((n * 2 + 255) / 256) * 256; return r; };
  auto allocF = [&](size_t n) { float* r = (float*)wp; wp += ((n * 4 + 255) / 256) * 256; return r; };

  unsigned short* WkT   = allocB(2048u * 256);
  unsigned short* WvT   = allocB(2048u * 256);
  unsigned short* WqT   = allocB((size_t)2048 * 2048);
  unsigned short* gWihB = allocB((size_t)16 * 384 * 128);
  unsigned short* gWhhB = allocB((size_t)16 * 384 * 128);
  unsigned short* mW1T  = allocB((size_t)16 * 128 * 128);
  unsigned short* mW2T  = allocB((size_t)16 * 128 * 128);
  unsigned short* cW1T  = allocB(512u * 128);
  unsigned short* cW2T  = allocB((size_t)512 * 512);
  unsigned short* cW3T  = allocB((size_t)512 * 512);
  unsigned short* cW4T  = allocB(128u * 512);
  unsigned short* cvecB = allocB(1024u * 128);
  unsigned short* xln   = allocB((size_t)32768 * 256);
  unsigned short* kbf   = allocB((size_t)32768 * 2048);
  unsigned short* vbf   = allocB((size_t)32768 * 2048);
  unsigned short* sbuf  = allocB(256u * 2048);     // reused: LN(slots), BLN(slots)
  unsigned short* qbf   = allocB(256u * 2048);
  unsigned short* attnT = allocB((size_t)32 * 8 * 1024);
  unsigned short* updB  = allocB(256u * 2048);
  unsigned short* slotsB= allocB(256u * 2048);
  unsigned short* h1bf  = allocB(256u * 2048);
  unsigned short* c1    = allocB((size_t)1024 * 512);
  unsigned short* c2    = allocB((size_t)1024 * 512);
  unsigned short* c3    = allocB((size_t)1024 * 512);
  unsigned short* memB  = allocB(64u * 2048);
  unsigned short* aB    = allocB((size_t)512 * 8 * 64);
  float* slotsF  = allocF(256u * 2048);
  float* dots    = allocF((size_t)32 * 1024 * 8);
  float* gi      = allocF(256u * 6144);
  float* gh      = allocF(256u * 6144);
  float* c4      = allocF(1024u * 128);
  float* scores  = allocF((size_t)512 * 8 * 64);

  float* attnOut = (float*)d_out + (size_t)32 * 8 * 2048;

  // wide: 128x128 block tile, 32x64 wave tile (8 wmma / 12 ds-loads per K-step)
  auto wide = [&](const GemmP& g, int zTot) {
    dim3 grid((g.N + 127) / 128, (g.M + 127) / 128, (unsigned)zTot);
    gemm_wmma_k<4, 2, 2, 4><<<grid, dim3(256), 0, stream>>>(g);
  };
  // narrow: 128x64 block tile for tiny-N (dots, N=8)
  auto narrow = [&](const GemmP& g, int zTot) {
    dim3 grid((g.N + 63) / 64, (g.M + 127) / 128, (unsigned)zTot);
    gemm_wmma_k<4, 2, 2, 2><<<grid, dim3(256), 0, stream>>>(g);
  };
  // skinny: 16x256 block tile for tiny-M (slot-side GEMMs, M=8)
  auto skinny = [&](const GemmP& g, int zTot) {
    dim3 grid((g.N + 255) / 256, (g.M + 15) / 16, (unsigned)zTot);
    gemm_wmma_k<1, 8, 1, 2><<<grid, dim3(256), 0, stream>>>(g);
  };
  auto grd = [](long n) { return dim3((unsigned)((n + 255) / 256)); };

  // ---- weight conversion / transposition (out x in, bf16) ----
  transpose_cvt_k<<<grd(524288), 256, 0, stream>>>(Wk, WkT, 256, 2048, 524288);
  transpose_cvt_k<<<grd(524288), 256, 0, stream>>>(Wv, WvT, 256, 2048, 524288);
  transpose_cvt_k<<<grd(4194304), 256, 0, stream>>>(Wq, WqT, 2048, 2048, 4194304);
  transpose_cvt_k<<<grd(262144), 256, 0, stream>>>(mW1, mW1T, 128, 128, 262144);
  transpose_cvt_k<<<grd(262144), 256, 0, stream>>>(mW2, mW2T, 128, 128, 262144);
  transpose_cvt_k<<<grd(65536), 256, 0, stream>>>(cW1, cW1T, 128, 512, 65536);
  transpose_cvt_k<<<grd(262144), 256, 0, stream>>>(cW2, cW2T, 512, 512, 262144);
  transpose_cvt_k<<<grd(262144), 256, 0, stream>>>(cW3, cW3T, 512, 512, 262144);
  transpose_cvt_k<<<grd(65536), 256, 0, stream>>>(cW4, cW4T, 512, 128, 65536);
  cvt_bf16_k<<<grd(786432), 256, 0, stream>>>(gWih, gWihB, 786432);   // already [blk][out][in]
  cvt_bf16_k<<<grd(786432), 256, 0, stream>>>(gWhh, gWhhB, 786432);
  cvt_bf16_k<<<grd(131072), 256, 0, stream>>>(cvec, cvecB, 131072);

  // ---- input LN + slots init ----
  ln_rows_k<<<32768, 256, 0, stream>>>(inputs, xln, nig, nib, 256);
  slots_init_k<<<grd(524288), 256, 0, stream>>>(mu, lsig, noise, slotsF, 524288);

  // ---- k, v projections: [32768,256] x [2048,256]^T -> bf16 [32768,2048] ----
  wide(mkgemm(xln, 256, 0, 0, WkT, 256, 0, 0, kbf, 2048, 0, 0, bk, 0, 0,
              32768, 2048, 256, 1, 1.f, GF_OUTBF16), 1);
  wide(mkgemm(xln, 256, 0, 0, WvT, 256, 0, 0, vbf, 2048, 0, 0, bv, 0, 0,
              32768, 2048, 256, 1, 1.f, GF_OUTBF16), 1);

  // ---- concept memory MLP (iteration-invariant) ----
  wide(mkgemm(cvecB, 128, 0, 0, cW1T, 128, 0, 0, c1, 512, 0, 0, cb1, 0, 0,
              1024, 512, 128, 1, 1.f, GF_RELU | GF_OUTBF16), 1);
  wide(mkgemm(c1, 512, 0, 0, cW2T, 512, 0, 0, c2, 512, 0, 0, cb2, 0, 0,
              1024, 512, 512, 1, 1.f, GF_RELU | GF_OUTBF16), 1);
  wide(mkgemm(c2, 512, 0, 0, cW3T, 512, 0, 0, c3, 512, 0, 0, cb3, 0, 0,
              1024, 512, 512, 1, 1.f, GF_RELU | GF_OUTBF16), 1);
  wide(mkgemm(c3, 512, 0, 0, cW4T, 512, 0, 0, c4, 128, 0, 0, cb4, 0, 0,
              1024, 128, 512, 1, 1.f, 0), 1);
  ln_rows_k<<<1024, 256, 0, stream>>>(c4, memB, nullptr, nullptr, 128);  // BLN -> mem [64,2048] bf16

  const float dotsScale = 1.f / sqrtf(2048.f);
  const float bwaScale  = 1.f / sqrtf(128.f);

  for (int it = 0; it < 3; ++it) {
    // s = LN(slots); q = s @ Wq + bq  (bf16)
    ln_rows_k<<<256, 256, 0, stream>>>(slotsF, sbuf, nsg, nsb, 2048);
    wide(mkgemm(sbuf, 2048, 0, 0, WqT, 2048, 0, 0, qbf, 2048, 0, 0, bq, 0, 0,
                256, 2048, 2048, 1, 1.f, GF_OUTBF16), 1);
    // dots[b,n,s] = scale * k . q   (batched over b)
    narrow(mkgemm(kbf, 2048, 0, (long)1024 * 2048, qbf, 2048, 0, (long)8 * 2048,
                  dots, 8, 0, 8192, nullptr, 0, 0,
                  1024, 8, 2048, 32, dotsScale, 0), 32);
    // softmax over inputs + EPS renorm -> attn (output) + attn^T bf16
    col_softmax_k<<<256, 256, 0, stream>>>(dots, attnOut, attnT);
    // updates[b,s,d] = attn^T @ v   (transB: v is [n][d]) -> bf16
    skinny(mkgemm(attnT, 1024, 0, 8192, vbf, 2048, 0, (long)1024 * 2048,
                  updB, 2048, 0, 16384, nullptr, 0, 0,
                  8, 2048, 1024, 32, 1.f, GF_TRANSB | GF_OUTBF16), 32);
    // GRU gates: gi = updates_blk @ gWih^T + bih ; gh = slots_blk @ gWhh^T + bhh
    cvt_bf16_k<<<grd(524288), 256, 0, stream>>>(slotsF, slotsB, 524288);
    wide(mkgemm(updB, 2048, 0, 128, gWihB, 128, 0, (long)384 * 128,
                gi, 6144, 0, 384, gbih, 0, 384,
                256, 384, 128, 16, 1.f, 0), 16);
    wide(mkgemm(slotsB, 2048, 0, 128, gWhhB, 128, 0, (long)384 * 128,
                gh, 6144, 0, 384, gbhh, 0, 384,
                256, 384, 128, 16, 1.f, 0), 16);
    gru_elem_k<<<grd(524288), 256, 0, stream>>>(gi, gh, slotsF);
    // block MLP residual: slots += blin2(relu(blin1(BLN(slots))))
    ln_rows_k<<<4096, 256, 0, stream>>>(slotsF, sbuf, nullptr, nullptr, 128);
    wide(mkgemm(sbuf, 2048, 0, 128, mW1T, 128, 0, (long)128 * 128,
                h1bf, 2048, 0, 128, mb1, 0, 128,
                256, 128, 128, 16, 1.f, GF_RELU | GF_OUTBF16), 16);
    wide(mkgemm(h1bf, 2048, 0, 128, mW2T, 128, 0, (long)128 * 128,
                slotsF, 2048, 0, 128, mb2, 0, 128,
                256, 128, 128, 16, 1.f, GF_ACC), 16);
    // block-wise attention over prototypes: scores = (BLN(slots)/sqrt(BD)) . mem
    ln_rows_k<<<4096, 256, 0, stream>>>(slotsF, sbuf, nullptr, nullptr, 128);
    skinny(mkgemm(sbuf, 2048, 16384, 128, memB, 2048, 0, 128,
                  scores, 64, 8192, 512, nullptr, 0, 0,
                  8, 64, 128, 16, bwaScale, 0), 512);
    softmax64_k<<<512, 256, 0, stream>>>(scores, aB);
    // slots = a @ mem  (transB: mem is [s][d])
    skinny(mkgemm(aB, 64, 8192, 512, memB, 2048, 0, 128,
                  slotsF, 2048, 16384, 128, nullptr, 0, 0,
                  8, 128, 64, 16, 1.f, GF_TRANSB), 512);
  }

  // output: (slots, attn) concatenated; attn already in place
  hipMemcpyAsync(d_out, slotsF, (size_t)524288 * sizeof(float),
                 hipMemcpyDeviceToDevice, stream);
}
